// EfficientSpatioTemporalBlock_49624052138591
// MI455X (gfx1250) — compile-verified, matrix operand order checked by reading, not run on hardware
//
#include <hip/hip_runtime.h>
#include <hip/hip_bf16.h>

// CDNA5 / gfx1250 wave32 WMMA types
typedef __attribute__((ext_vector_type(16))) __bf16 v16bf;
typedef __attribute__((ext_vector_type(8)))  __bf16 v8bf;
typedef __attribute__((ext_vector_type(8)))  float  v8f;

#define B_   4
#define CIN  64
#define CH   256
#define T_   16
#define H_   64
#define W_   64
#define SP   (H_*W_)     /* 4096  */
#define P_   (T_*SP)     /* 65536 */
#define EPS_ 1e-5f

// Drain the async-to-LDS counter (ASYNCcnt).
__device__ __forceinline__ void wait_async0() {
#if __has_builtin(__builtin_amdgcn_s_wait_asynccnt)
    __builtin_amdgcn_s_wait_asynccnt(0);
#else
    asm volatile("s_wait_asynccnt 0x0" ::: "memory");
#endif
}

// One per-lane 16-byte async copy global -> LDS (CDNA5 GLOBAL_LOAD_ASYNC_TO_LDS_B128).
__device__ __forceinline__ void async_copy_b128(const void* gsrc, void* lds_dst) {
    unsigned int  laddr = (unsigned int)(size_t)lds_dst;   // LDS aperture: low 32 bits
    unsigned long long gaddr = (unsigned long long)(size_t)gsrc;
    asm volatile("global_load_async_to_lds_b128 %0, %1, off"
                 :: "v"(laddr), "v"(gaddr) : "memory");
}

// ---------------------------------------------------------------------------
// Kernel A: expand 1x1 conv as bf16 WMMA GEMM  h1[b,c,p] = sum_k w1[c,k]*x[b,k,p]
// Block tile: M=64 x N=128, K=64. 8 waves = 4(M) x 2(N-half); 8 v_wmma per wave.
// Per-(b,c) sum/sumsq for InstanceNorm1 via shuffle+atomics.
// Output staged through LDS so global stores are coalesced b128.
// ---------------------------------------------------------------------------
__global__ __launch_bounds__(256)
void k_expand_wmma(const float* __restrict__ x, const float* __restrict__ w1,
                   __bf16* __restrict__ h1,
                   float* __restrict__ s_sum, float* __restrict__ s_sq)
{
    __shared__ __align__(32) __bf16 wA[64][64];     //  8 KB  [M][K]
    __shared__ __align__(32) __bf16 xBt[128][64];   // 16 KB  [N][K] (transposed)
    __bf16* outT = &xBt[0][0];                      // reuse as [64][128] output tile

    const int tid = threadIdx.x;
    const int b   = blockIdx.z;
    const int c0  = blockIdx.y * 64;
    const int p0  = blockIdx.x * 128;

    for (int i = tid; i < 64 * 64 / 4; i += 256) {  // float4 weight loads
        int m = i >> 4, k4 = (i & 15) * 4;
        float4 v = *(const float4*)&w1[(c0 + m) * 64 + k4];
        wA[m][k4 + 0] = (__bf16)v.x; wA[m][k4 + 1] = (__bf16)v.y;
        wA[m][k4 + 2] = (__bf16)v.z; wA[m][k4 + 3] = (__bf16)v.w;
    }
    for (int i = tid; i < 64 * 128 / 4; i += 256) { // float4 activation loads
        int k = i >> 5, n4 = (i & 31) * 4;
        float4 v = *(const float4*)&x[((size_t)(b * CIN + k)) * P_ + p0 + n4];
        xBt[n4 + 0][k] = (__bf16)v.x; xBt[n4 + 1][k] = (__bf16)v.y;
        xBt[n4 + 2][k] = (__bf16)v.z; xBt[n4 + 3][k] = (__bf16)v.w;
    }
    __syncthreads();

    const int wid = tid >> 5, lane = tid & 31;
    const int lr = lane & 15, lh = lane >> 4;
    const int m0w = (wid >> 1) * 16;
    const int n0w = (wid & 1) * 64;

    v8f acc[4] = {};
#pragma unroll
    for (int kc = 0; kc < 64; kc += 32) {
        v16bf a = *(const v16bf*)&wA[m0w + lr][kc + lh * 16];
#pragma unroll
        for (int j = 0; j < 4; ++j) {
            v16bf bb = *(const v16bf*)&xBt[n0w + j * 16 + lr][kc + lh * 16];
            acc[j] = __builtin_amdgcn_wmma_f32_16x16x32_bf16(
                false, a, false, bb, (short)0, acc[j], false, false);
        }
    }

    // stats (per-channel sum / sumsq over the 16 N-lanes, then atomics)
    float fsum[8] = {}, fsq[8] = {};
#pragma unroll
    for (int j = 0; j < 4; ++j)
#pragma unroll
        for (int r = 0; r < 8; ++r) {
            float v = acc[j][r];
            fsum[r] += v; fsq[r] += v * v;
        }
#pragma unroll
    for (int r = 0; r < 8; ++r) {
        float s = fsum[r], q = fsq[r];
        for (int m = 1; m < 16; m <<= 1) { s += __shfl_xor(s, m); q += __shfl_xor(q, m); }
        if (lr == 0) {
            int c = c0 + m0w + r + 8 * lh;
            atomicAdd(&s_sum[b * CH + c], s);
            atomicAdd(&s_sq[b * CH + c], q);
        }
    }

    // scatter D tiles to LDS, then coalesced b128 global stores
    __syncthreads();
#pragma unroll
    for (int j = 0; j < 4; ++j)
#pragma unroll
        for (int r = 0; r < 8; ++r) {
            int m = m0w + r + 8 * lh;            // ISA C/D layout: M = r + 8*(lane>=16)
            int n = n0w + j * 16 + lr;           // N = lane & 15
            outT[m * 128 + n] = (__bf16)acc[j][r];
        }
    __syncthreads();
    for (int i = tid; i < 64 * 128 / 8; i += 256) {
        int m = i >> 4, n8 = (i & 15) * 8;
        *(v8bf*)&h1[((size_t)(b * CH + c0 + m)) * P_ + p0 + n8] =
            *(const v8bf*)&outT[m * 128 + n8];
    }
}

// ---------------------------------------------------------------------------
// Finalize InstanceNorm stats: mean, rsqrt(var+eps)
// ---------------------------------------------------------------------------
__global__ void k_finalize(const float* __restrict__ sum, const float* __restrict__ sq,
                           float* __restrict__ mean, float* __restrict__ rs,
                           int n, float invN)
{
    int i = blockIdx.x * 256 + threadIdx.x;
    if (i < n) {
        float m = sum[i] * invN;
        float v = sq[i] * invN - m * m;
        mean[i] = m;
        rs[i] = rsqrtf(v + EPS_);
    }
}

// ---------------------------------------------------------------------------
// Kernel C: IN1+ReLU+TIM, depthwise 3x3 spatial conv (LDS halo tile), stats2
// TIM: c<32 -> read t+1 (zero at t==15); 32<=c<64 -> zero; else identity.
// ---------------------------------------------------------------------------
__global__ __launch_bounds__(256)
void k_tim_dws(const __bf16* __restrict__ h1, const float* __restrict__ wds,
               const float* __restrict__ m1, const float* __restrict__ rs1,
               __bf16* __restrict__ h2,
               float* __restrict__ s_sum, float* __restrict__ s_sq)
{
    __shared__ float gt[66][66];
    __shared__ float red[256];
    const int tid = threadIdx.x;
    const int bc  = blockIdx.x;           // b*CH + c
    const int c   = bc & 255;
    const int t   = blockIdx.y;

    const bool zeroed = (c >= 32 && c < 64);
    const int  tsrc   = (c < 32) ? t + 1 : t;
    const bool valid  = (!zeroed) && (tsrc < T_);
    const float mm = m1[bc], rr = rs1[bc];
    const size_t base = ((size_t)bc) * P_ + (size_t)tsrc * SP;

    for (int i = tid; i < 66 * 66; i += 256) {
        int hh = (i / 66) - 1, ww = (i % 66) - 1;
        float v = 0.f;
        if (valid && hh >= 0 && hh < H_ && ww >= 0 && ww < W_) {
            float r = ((float)h1[base + hh * W_ + ww] - mm) * rr;
            v = fmaxf(r, 0.f);
        }
        gt[i / 66][i % 66] = v;
    }
    __syncthreads();

    float w[9];
#pragma unroll
    for (int q = 0; q < 9; ++q) w[q] = wds[c * 9 + q];

    float lsum = 0.f, lsq = 0.f;
    const size_t obase = ((size_t)bc) * P_ + (size_t)t * SP;
    for (int i0 = tid * 8; i0 < SP; i0 += 256 * 8) {   // 8 outputs per iter -> b128 store
        int hh = i0 >> 6, wb = i0 & 63;
        v8bf o8;
#pragma unroll
        for (int u = 0; u < 8; ++u) {
            float a = 0.f;
#pragma unroll
            for (int kh = 0; kh < 3; ++kh)
#pragma unroll
                for (int kw = 0; kw < 3; ++kw)
                    a = fmaf(gt[hh + kh][wb + u + kw], w[kh * 3 + kw], a);
            o8[u] = (__bf16)a;
            lsum += a; lsq += a * a;
        }
        *(v8bf*)&h2[obase + i0] = o8;
    }
    red[tid] = lsum; __syncthreads();
    for (int s = 128; s > 0; s >>= 1) { if (tid < s) red[tid] += red[tid + s]; __syncthreads(); }
    float bs = red[0]; __syncthreads();
    red[tid] = lsq; __syncthreads();
    for (int s = 128; s > 0; s >>= 1) { if (tid < s) red[tid] += red[tid + s]; __syncthreads(); }
    if (tid == 0) { atomicAdd(&s_sum[bc], bs); atomicAdd(&s_sq[bc], red[0]); }
}

// ---------------------------------------------------------------------------
// Kernel E: IN2+ReLU, depthwise 3-tap temporal conv, stats3.
// The three raw bf16 time-slices are staged into LDS with the CDNA5 async
// data mover (GLOBAL_LOAD_ASYNC_TO_LDS_B128, drained via s_wait_asynccnt);
// normalization/ReLU happen per element in registers afterwards.
// ---------------------------------------------------------------------------
__global__ __launch_bounds__(256)
void k_dwt(const __bf16* __restrict__ h2, const float* __restrict__ wdt,
           const float* __restrict__ m2, const float* __restrict__ rs2,
           __bf16* __restrict__ h3,
           float* __restrict__ s_sum, float* __restrict__ s_sq)
{
    __shared__ __align__(16) __bf16 sl[3][SP];   // 24 KB: t-1, t, t+1 slices
    __shared__ float red[256];
    const int tid = threadIdx.x;
    const int bc  = blockIdx.x;
    const int c   = bc & 255;
    const int t   = blockIdx.y;
    const float mm = m2[bc], rr = rs2[bc];
    const float w0 = wdt[c * 3 + 0], w1 = wdt[c * 3 + 1], w2 = wdt[c * 3 + 2];
    const size_t pbase = ((size_t)bc) * P_;
    const size_t sbase = pbase + (size_t)t * SP;
    const bool hasP = (t > 0), hasN = (t < T_ - 1);

    // async-stage the valid slices (512 x 16B chunks each; 2 chunks/thread)
#pragma unroll
    for (int i = tid; i < SP / 8; i += 256) {
        async_copy_b128(&h2[sbase + i * 8], &sl[1][i * 8]);
        if (hasP) async_copy_b128(&h2[sbase - SP + i * 8], &sl[0][i * 8]);
        if (hasN) async_copy_b128(&h2[sbase + SP + i * 8], &sl[2][i * 8]);
    }
    wait_async0();
    __syncthreads();

    float lsum = 0.f, lsq = 0.f;
    for (int i0 = tid * 8; i0 < SP; i0 += 256 * 8) {
        v8bf c8 = *(const v8bf*)&sl[1][i0];
        v8bf p8 = {}, n8 = {};
        if (hasP) p8 = *(const v8bf*)&sl[0][i0];
        if (hasN) n8 = *(const v8bf*)&sl[2][i0];
        v8bf o8;
#pragma unroll
        for (int u = 0; u < 8; ++u) {
            float cur = fmaxf(((float)c8[u] - mm) * rr, 0.f);
            float prv = hasP ? fmaxf(((float)p8[u] - mm) * rr, 0.f) : 0.f;
            float nxt = hasN ? fmaxf(((float)n8[u] - mm) * rr, 0.f) : 0.f;
            float a = w0 * prv + w1 * cur + w2 * nxt;
            o8[u] = (__bf16)a;
            lsum += a; lsq += a * a;
        }
        *(v8bf*)&h3[sbase + i0] = o8;
    }
    red[tid] = lsum; __syncthreads();
    for (int s = 128; s > 0; s >>= 1) { if (tid < s) red[tid] += red[tid + s]; __syncthreads(); }
    float bs = red[0]; __syncthreads();
    red[tid] = lsq; __syncthreads();
    for (int s = 128; s > 0; s >>= 1) { if (tid < s) red[tid] += red[tid + s]; __syncthreads(); }
    if (tid == 0) { atomicAdd(&s_sum[bc], bs); atomicAdd(&s_sq[bc], red[0]); }
}

// ---------------------------------------------------------------------------
// Kernel G: SE pooling  ysum[b,c] += sum_p relu((h3-m3)*rs3)  (v8bf streaming)
// ---------------------------------------------------------------------------
__global__ __launch_bounds__(256)
void k_sepool(const __bf16* __restrict__ h3, const float* __restrict__ m3,
              const float* __restrict__ rs3, float* __restrict__ ysum)
{
    __shared__ float red[256];
    const int tid = threadIdx.x;
    const int bc  = blockIdx.x;
    const int t   = blockIdx.y;
    const float mm = m3[bc], rr = rs3[bc];
    const size_t base = ((size_t)bc) * P_ + (size_t)t * SP;

    float lsum = 0.f;
    for (int i0 = tid * 8; i0 < SP; i0 += 256 * 8) {
        v8bf v8 = *(const v8bf*)&h3[base + i0];
#pragma unroll
        for (int u = 0; u < 8; ++u)
            lsum += fmaxf(((float)v8[u] - mm) * rr, 0.f);
    }
    red[tid] = lsum; __syncthreads();
    for (int s = 128; s > 0; s >>= 1) { if (tid < s) red[tid] += red[tid + s]; __syncthreads(); }
    if (tid == 0) atomicAdd(&ysum[bc], red[0]);
}

// ---------------------------------------------------------------------------
// Kernel H: SE MLP 256 -> 64 (ReLU) -> 256 (sigmoid). Tiny: one block.
// ---------------------------------------------------------------------------
__global__ __launch_bounds__(256)
void k_semlp(const float* __restrict__ ysum, const float* __restrict__ wse1,
             const float* __restrict__ wse2, float* __restrict__ se)
{
    __shared__ float ym[256];
    __shared__ float y1[64];
    const int tid = threadIdx.x;
    for (int b = 0; b < B_; ++b) {
        ym[tid] = ysum[b * CH + tid] * (1.0f / (float)P_);
        __syncthreads();
        if (tid < 64) {
            float a = 0.f;
            for (int cc = 0; cc < 256; ++cc) a = fmaf(wse1[tid * 256 + cc], ym[cc], a);
            y1[tid] = fmaxf(a, 0.f);
        }
        __syncthreads();
        {
            float a = 0.f;
            for (int o = 0; o < 64; ++o) a = fmaf(wse2[tid * 64 + o], y1[o], a);
            se[b * CH + tid] = 1.0f / (1.0f + expf(-a));
        }
        __syncthreads();
    }
}

// ---------------------------------------------------------------------------
// Kernel I: proj 1x1 GEMM with SE folded into weights, bf16 WMMA, K=256.
// proj[b,o,p] = sum_c relu((h3-m3)*rs3)[b,c,p] * (w_proj[o,c]*se[b,c])
// Block tile 64(M) x 128(N); K in 4 staged chunks -> 32 v_wmma per wave.
// Output staged through LDS (reusing wA) for coalesced float4 stores.
// ---------------------------------------------------------------------------
__global__ __launch_bounds__(256)
void k_proj_wmma(const __bf16* __restrict__ h3, const float* __restrict__ wproj,
                 const float* __restrict__ se, const float* __restrict__ m3,
                 const float* __restrict__ rs3, float* __restrict__ proj,
                 float* __restrict__ s_sum, float* __restrict__ s_sq)
{
    __shared__ __align__(32) __bf16 wA[64][256];    // 32 KB [M][K]
    __shared__ __align__(32) __bf16 xBt[128][64];   // 16 KB [N][Kchunk]
    float* outF = (float*)&wA[0][0];                // reuse as [64][128] f32 out tile

    const int tid = threadIdx.x;
    const int b   = blockIdx.z;
    const int p0  = blockIdx.x * 128;

    for (int i = tid; i < 64 * 256 / 4; i += 256) { // float4 weight loads
        int m = i >> 6, c4 = (i & 63) * 4;
        float4 v = *(const float4*)&wproj[m * 256 + c4];
        wA[m][c4 + 0] = (__bf16)(v.x * se[b * CH + c4 + 0]);
        wA[m][c4 + 1] = (__bf16)(v.y * se[b * CH + c4 + 1]);
        wA[m][c4 + 2] = (__bf16)(v.z * se[b * CH + c4 + 2]);
        wA[m][c4 + 3] = (__bf16)(v.w * se[b * CH + c4 + 3]);
    }

    const int wid = tid >> 5, lane = tid & 31;
    const int lr = lane & 15, lh = lane >> 4;
    const int m0w = (wid >> 1) * 16;
    const int n0w = (wid & 1) * 64;

    v8f acc[4] = {};
#pragma unroll
    for (int kc0 = 0; kc0 < 256; kc0 += 64) {
        __syncthreads();
        for (int i = tid; i < 64 * 128 / 8; i += 256) {   // v8bf activation loads
            int k = i >> 4, n8 = (i & 15) * 8;
            int cc = kc0 + k;
            const float m3c = m3[b * CH + cc], r3c = rs3[b * CH + cc];
            v8bf v8 = *(const v8bf*)&h3[((size_t)(b * CH + cc)) * P_ + p0 + n8];
#pragma unroll
            for (int u = 0; u < 8; ++u)
                xBt[n8 + u][k] = (__bf16)fmaxf(((float)v8[u] - m3c) * r3c, 0.f);
        }
        __syncthreads();
#pragma unroll
        for (int kc = 0; kc < 64; kc += 32) {
            v16bf a = *(const v16bf*)&wA[m0w + lr][kc0 + kc + lh * 16];
#pragma unroll
            for (int j = 0; j < 4; ++j) {
                v16bf bb = *(const v16bf*)&xBt[n0w + j * 16 + lr][kc + lh * 16];
                acc[j] = __builtin_amdgcn_wmma_f32_16x16x32_bf16(
                    false, a, false, bb, (short)0, acc[j], false, false);
            }
        }
    }

    // stats
    float fsum[8] = {}, fsq[8] = {};
#pragma unroll
    for (int j = 0; j < 4; ++j)
#pragma unroll
        for (int r = 0; r < 8; ++r) {
            float v = acc[j][r];
            fsum[r] += v; fsq[r] += v * v;
        }
#pragma unroll
    for (int r = 0; r < 8; ++r) {
        float s = fsum[r], q = fsq[r];
        for (int m = 1; m < 16; m <<= 1) { s += __shfl_xor(s, m); q += __shfl_xor(q, m); }
        if (lr == 0) {
            int o = m0w + r + 8 * lh;
            atomicAdd(&s_sum[b * 64 + o], s);
            atomicAdd(&s_sq[b * 64 + o], q);
        }
    }

    // scatter D tiles to LDS (over wA), then coalesced float4 stores
    __syncthreads();
#pragma unroll
    for (int j = 0; j < 4; ++j)
#pragma unroll
        for (int r = 0; r < 8; ++r) {
            int m = m0w + r + 8 * lh;
            int n = n0w + j * 16 + lr;
            outF[m * 128 + n] = acc[j][r];
        }
    __syncthreads();
    for (int i = tid; i < 64 * 128 / 4; i += 256) {
        int m = i >> 5, n4 = (i & 31) * 4;
        *(float4*)&proj[((size_t)(b * 64 + m)) * P_ + p0 + n4] =
            *(const float4*)&outF[m * 128 + n4];
    }
}

// ---------------------------------------------------------------------------
// Kernel K: IN4 + residual + 2x2 spatial maxpool -> out[4,64,16,32,32]
// ---------------------------------------------------------------------------
__global__ __launch_bounds__(256)
void k_residual_pool(const float* __restrict__ proj, const float* __restrict__ x,
                     const float* __restrict__ m4, const float* __restrict__ rs4,
                     float* __restrict__ out)
{
    int i = blockIdx.x * 256 + threadIdx.x;     // 4*64*16*32*32 = 4194304 outputs
    if (i >= 4194304) return;
    int ow = i & 31;
    int oh = (i >> 5) & 31;
    int t  = (i >> 10) & 15;
    int oc = (i >> 14) & 63;
    int b  = i >> 20;
    const float mm = m4[b * 64 + oc], rr = rs4[b * 64 + oc];
    const size_t base = ((size_t)(b * 64 + oc)) * P_ + (size_t)t * SP;
    const size_t r0 = base + (size_t)(2 * oh) * W_ + 2 * ow;
    const size_t r1 = r0 + W_;
    float2 pa = *(const float2*)&proj[r0];
    float2 pb = *(const float2*)&proj[r1];
    float2 xa = *(const float2*)&x[r0];
    float2 xb = *(const float2*)&x[r1];
    float v0 = (pa.x - mm) * rr + xa.x;
    float v1 = (pa.y - mm) * rr + xa.y;
    float v2 = (pb.x - mm) * rr + xb.x;
    float v3 = (pb.y - mm) * rr + xb.y;
    out[i] = fmaxf(fmaxf(v0, v1), fmaxf(v2, v3));
}

// ---------------------------------------------------------------------------
// Launch: workspace layout (bytes)
//   [0,128MB)      buf1 : h1 bf16, later reused for h3 bf16
//   [128MB,256MB)  buf2 : h2 bf16, later reused for proj f32 (64 MB)
//   [256MB,+64KB)  stats region (f32)
// ---------------------------------------------------------------------------
extern "C" void kernel_launch(void* const* d_in, const int* in_sizes, int n_in,
                              void* d_out, int out_size, void* d_ws, size_t ws_size,
                              hipStream_t stream)
{
    (void)in_sizes; (void)n_in; (void)out_size; (void)ws_size;
    const float* x     = (const float*)d_in[0];
    const float* w1    = (const float*)d_in[1];
    const float* wds   = (const float*)d_in[2];
    const float* wdt   = (const float*)d_in[3];
    const float* wse1  = (const float*)d_in[4];
    const float* wse2  = (const float*)d_in[5];
    const float* wproj = (const float*)d_in[6];
    float* out = (float*)d_out;

    char* ws = (char*)d_ws;
    __bf16* buf1 = (__bf16*)ws;                                   // h1 / h3
    __bf16* buf2 = (__bf16*)(ws + (size_t)128 * 1024 * 1024);     // h2
    float*  proj = (float*)buf2;                                  // proj overlays h2
    float*  st   = (float*)(ws + (size_t)256 * 1024 * 1024);

    float* s1_sum = st + 0,     *s1_sq = st + 1024,  *s1_m = st + 2048,  *s1_r = st + 3072;
    float* s2_sum = st + 4096,  *s2_sq = st + 5120,  *s2_m = st + 6144,  *s2_r = st + 7168;
    float* s3_sum = st + 8192,  *s3_sq = st + 9216,  *s3_m = st + 10240, *s3_r = st + 11264;
    float* s4_sum = st + 12288, *s4_sq = st + 12544, *s4_m = st + 12800, *s4_r = st + 13056;
    float* ysum   = st + 13312;                                   // [1024]
    float* se     = st + 14336;                                   // [1024]

    // zero the atomic accumulators each call (graph-capturable)
    hipMemsetAsync(st, 0, 14336 * sizeof(float), stream);

    k_expand_wmma<<<dim3(P_ / 128, CH / 64, B_), 256, 0, stream>>>(x, w1, buf1, s1_sum, s1_sq);
    k_finalize<<<4, 256, 0, stream>>>(s1_sum, s1_sq, s1_m, s1_r, 1024, 1.0f / (float)P_);
    k_tim_dws<<<dim3(B_ * CH, T_), 256, 0, stream>>>(buf1, wds, s1_m, s1_r, buf2, s2_sum, s2_sq);
    k_finalize<<<4, 256, 0, stream>>>(s2_sum, s2_sq, s2_m, s2_r, 1024, 1.0f / (float)P_);
    k_dwt<<<dim3(B_ * CH, T_), 256, 0, stream>>>(buf2, wdt, s2_m, s2_r, buf1, s3_sum, s3_sq);
    k_finalize<<<4, 256, 0, stream>>>(s3_sum, s3_sq, s3_m, s3_r, 1024, 1.0f / (float)P_);
    k_sepool<<<dim3(B_ * CH, T_), 256, 0, stream>>>(buf1, s3_m, s3_r, ysum);
    k_semlp<<<1, 256, 0, stream>>>(ysum, wse1, wse2, se);
    k_proj_wmma<<<dim3(P_ / 128, 1, B_), 256, 0, stream>>>(buf1, wproj, se, s3_m, s3_r,
                                                           proj, s4_sum, s4_sq);
    k_finalize<<<1, 256, 0, stream>>>(s4_sum, s4_sq, s4_m, s4_r, 256, 1.0f / (float)P_);
    k_residual_pool<<<4194304 / 256, 256, 0, stream>>>(proj, x, s4_m, s4_r, out);
}
